// MPNNFlockingModel_53644141527377
// MI455X (gfx1250) — matrix operand order repaired and checked by reading, not compiled, not run on hardware
//
#include <hip/hip_runtime.h>
#include <hip/hip_bf16.h>

typedef __attribute__((ext_vector_type(16))) _Float16 v16h;
typedef __attribute__((ext_vector_type(8)))  float    v8f;

#define WARP 32

// ---------------------------------------------------------------------------
// A fragment = W^T (M = out-channel, K = in-feature), bias folded at k == K.
// A 16x32 k order used here (contiguous, k = 16h + e): lanes 0-15 row M=lane, elem e -> K=e (0..15);
// lanes 16-31 row M=lane-16, elem e -> K=16+e (16..31).
// W is row-major [K][O] (numpy x @ W convention).
// ---------------------------------------------------------------------------
__device__ __forceinline__ v16h make_wfragA(const float* __restrict__ W,
                                            const float* __restrict__ Bias,
                                            int K, int O) {
  v16h a = {};
  int lane = threadIdx.x & (WARP - 1);
  int m = lane & 15;
  int klo = (lane < 16) ? 0 : 16;
#pragma unroll
  for (int e = 0; e < 16; ++e) {
    int k = klo + e;
    float v = 0.0f;
    if (m < O) {
      if (k < K)       v = W[k * O + m];
      else if (k == K) v = Bias[m];
    }
    a[e] = (_Float16)v;
  }
  return a;
}

// B fragment = features^T (N = lane = row index, K = feature). B 32x16, same contiguous k order as A:
// lanes 0-15 col N=lane hold K=0..15 (elem e -> K=e); lanes 16-31 hold K=16..31
// (all zero padding here). Constant 1.0 at k == nf multiplies the bias column.
__device__ __forceinline__ v16h make_bfrag(const float* __restrict__ f, int nf,
                                           bool valid) {
  v16h b = {};
  int lane = threadIdx.x & (WARP - 1);
  bool lo = (lane < 16) && valid;
#pragma unroll
  for (int e = 0; e < 16; ++e) {
    float v = 0.0f;
    if (lo) {
      if (e < nf)       v = f[e];
      else if (e == nf) v = 1.0f;
    }
    b[e] = (_Float16)v;
  }
  return b;
}

__device__ __forceinline__ v8f wmma16(v16h A, v16h B) {
  v8f c = {};
  return __builtin_amdgcn_wmma_f32_16x16x32_f16(false, A, false, B,
                                                (short)0, c, false, false);
}

// Wave shuffle reduce + per-block LDS reduce -> partial array [block][2*NCH].
template <int NCH>
__device__ __forceinline__ void stats_reduce_store(float* s, float* q,
                                                   float* __restrict__ part) {
#pragma unroll
  for (int off = 16; off >= 1; off >>= 1) {
#pragma unroll
    for (int r = 0; r < NCH; ++r) {
      s[r] += __shfl_xor(s[r], off, WARP);
      q[r] += __shfl_xor(q[r], off, WARP);
    }
  }
  __shared__ float bs[2 * NCH];
  if ((int)threadIdx.x < 2 * NCH) bs[threadIdx.x] = 0.0f;
  __syncthreads();
  if ((threadIdx.x & (WARP - 1)) == 0) {
#pragma unroll
    for (int r = 0; r < NCH; ++r) {
      atomicAdd(&bs[r], s[r]);
      atomicAdd(&bs[NCH + r], q[r]);
    }
  }
  __syncthreads();
  if ((int)threadIdx.x < 2 * NCH)
    part[blockIdx.x * 2 * NCH + threadIdx.x] = bs[threadIdx.x];
}

// Gather one edge's 8 features: [h[dst](4), h[src](4)].
__device__ __forceinline__ bool load_edge_feats(const int* __restrict__ ei,
                                                const float* __restrict__ h,
                                                int E, int e, bool lanelo,
                                                float* f, int* dst_out,
                                                bool* eq_out) {
  bool valid = lanelo && (e < E);
  int dst = 0;
  bool eq = false;
#pragma unroll
  for (int r = 0; r < 8; ++r) f[r] = 0.0f;
  if (valid) {
    int src = ei[e];
    dst = ei[E + e];
    float4 hi = ((const float4*)h)[dst];
    float4 hj = ((const float4*)h)[src];
    f[0] = hi.x; f[1] = hi.y; f[2] = hi.z; f[3] = hi.w;
    f[4] = hj.x; f[5] = hj.y; f[6] = hj.z; f[7] = hj.w;
    eq = (hi.x == hj.x) && (hi.y == hj.y) && (hi.z == hj.z) && (hi.w == hj.w);
  }
  *dst_out = dst;
  *eq_out = eq;
  return valid;
}

// ---------------------------------------------------------------------------
__global__ void zero_kernel(float* __restrict__ p, int n) {
  for (int i = blockIdx.x * blockDim.x + threadIdx.x; i < n;
       i += gridDim.x * blockDim.x)
    p[i] = 0.0f;
}

__global__ void compute_h(const float* __restrict__ pos,
                          const float* __restrict__ vel, int N,
                          const float* __restrict__ W,
                          const float* __restrict__ B, float* __restrict__ h) {
  int n = blockIdx.x * blockDim.x + threadIdx.x;
  if (n >= N) return;
  float x0 = pos[2 * n], x1 = pos[2 * n + 1];
  float x2 = vel[2 * n], x3 = vel[2 * n + 1];
#pragma unroll
  for (int o = 0; o < 4; ++o)
    h[n * 4 + o] =
        B[o] + x0 * W[o] + x1 * W[4 + o] + x2 * W[8 + o] + x3 * W[12 + o];
}

// Edge pass A: x1 stats (8 ch over E rows)
__global__ void __launch_bounds__(256)
edge_pass1(const int* __restrict__ ei, const float* __restrict__ h, int E,
           const float* __restrict__ W1, const float* __restrict__ b1,
           float* __restrict__ part) {
  v16h A1 = make_wfragA(W1, b1, 8, 8);
  int lane = threadIdx.x & 31;
  int wave = (blockIdx.x * blockDim.x + threadIdx.x) >> 5;
  int nwaves = (gridDim.x * blockDim.x) >> 5;
  int ntiles = (E + 15) >> 4;
  float s[8], q[8];
#pragma unroll
  for (int r = 0; r < 8; ++r) { s[r] = 0.0f; q[r] = 0.0f; }
  for (int t = wave; t < ntiles; t += nwaves) {
    int e = t * 16 + (lane & 15);
    float f[8]; int dst; bool eq;
    bool valid = load_edge_feats(ei, h, E, e, lane < 16, f, &dst, &eq);
    v16h B = make_bfrag(f, 8, valid);
    v8f x1 = wmma16(A1, B);
    if (valid) {
#pragma unroll
      for (int r = 0; r < 8; ++r) { s[r] += x1[r]; q[r] += x1[r] * x1[r]; }
    }
  }
  stats_reduce_store<8>(s, q, part);
}

// Edge pass B: x2 stats (4 ch over E rows)
__global__ void __launch_bounds__(256)
edge_pass2(const int* __restrict__ ei, const float* __restrict__ h, int E,
           const float* __restrict__ W1, const float* __restrict__ b1,
           const float* __restrict__ W2, const float* __restrict__ b2,
           const float* __restrict__ bn1, float* __restrict__ part) {
  v16h A1 = make_wfragA(W1, b1, 8, 8);
  v16h A2 = make_wfragA(W2, b2, 8, 4);
  float a1[8], c1[8];
#pragma unroll
  for (int r = 0; r < 8; ++r) { a1[r] = bn1[r]; c1[r] = bn1[8 + r]; }
  int lane = threadIdx.x & 31;
  int wave = (blockIdx.x * blockDim.x + threadIdx.x) >> 5;
  int nwaves = (gridDim.x * blockDim.x) >> 5;
  int ntiles = (E + 15) >> 4;
  float s[4], q[4];
#pragma unroll
  for (int r = 0; r < 4; ++r) { s[r] = 0.0f; q[r] = 0.0f; }
  for (int t = wave; t < ntiles; t += nwaves) {
    int e = t * 16 + (lane & 15);
    float f[8]; int dst; bool eq;
    bool valid = load_edge_feats(ei, h, E, e, lane < 16, f, &dst, &eq);
    v16h B = make_bfrag(f, 8, valid);
    v8f x1 = wmma16(A1, B);
    float t1[8];
#pragma unroll
    for (int r = 0; r < 8; ++r) t1[r] = tanhf(a1[r] * x1[r] + c1[r]);
    v16h B2 = make_bfrag(t1, 8, valid);
    v8f x2 = wmma16(A2, B2);
    if (valid) {
#pragma unroll
      for (int r = 0; r < 4; ++r) { s[r] += x2[r]; q[r] += x2[r] * x2[r]; }
    }
  }
  stats_reduce_store<4>(s, q, part);
}

// Edge pass C: full message + scatter
__global__ void __launch_bounds__(256)
edge_pass3(const int* __restrict__ ei, const float* __restrict__ h, int E,
           const float* __restrict__ W1, const float* __restrict__ b1,
           const float* __restrict__ W2, const float* __restrict__ b2,
           const float* __restrict__ bn1, const float* __restrict__ bn2,
           float* __restrict__ aggA, float* __restrict__ aggM,
           float* __restrict__ cnt) {
  v16h A1 = make_wfragA(W1, b1, 8, 8);
  v16h A2 = make_wfragA(W2, b2, 8, 4);
  float a1[8], c1[8], a2[4], c2[4];
#pragma unroll
  for (int r = 0; r < 8; ++r) { a1[r] = bn1[r]; c1[r] = bn1[8 + r]; }
#pragma unroll
  for (int r = 0; r < 4; ++r) { a2[r] = bn2[r]; c2[r] = bn2[4 + r]; }
  int lane = threadIdx.x & 31;
  int wave = (blockIdx.x * blockDim.x + threadIdx.x) >> 5;
  int nwaves = (gridDim.x * blockDim.x) >> 5;
  int ntiles = (E + 15) >> 4;
  for (int t = wave; t < ntiles; t += nwaves) {
    int e = t * 16 + (lane & 15);
    float f[8]; int dst; bool eq;
    bool valid = load_edge_feats(ei, h, E, e, lane < 16, f, &dst, &eq);
    v16h B = make_bfrag(f, 8, valid);
    v8f x1 = wmma16(A1, B);
    float t1[8];
#pragma unroll
    for (int r = 0; r < 8; ++r) t1[r] = tanhf(a1[r] * x1[r] + c1[r]);
    v16h B2 = make_bfrag(t1, 8, valid);
    v8f x2 = wmma16(A2, B2);
    if (valid) {
      float m[4];
#pragma unroll
      for (int r = 0; r < 4; ++r) {
        m[r] = tanhf(a2[r] * x2[r] + c2[r]);
        if (eq) m[r] = 0.0f;
      }
      atomicAdd(&aggA[2 * dst + 0], m[0]);
      atomicAdd(&aggA[2 * dst + 1], m[1]);
      atomicAdd(&aggM[2 * dst + 0], m[2]);
      atomicAdd(&aggM[2 * dst + 1], m[3]);
      atomicAdd(&cnt[dst], 1.0f);
    }
  }
}

// Node pass 1: y1 = cat(h, aggr) @ upd_w1 + b1, store + stats (8 ch over N)
__global__ void __launch_bounds__(256)
node_pass1(const float* __restrict__ h, const float* __restrict__ aggA,
           const float* __restrict__ aggM, const float* __restrict__ cnt,
           int N, const float* __restrict__ W, const float* __restrict__ B,
           float* __restrict__ y1, float* __restrict__ part) {
  v16h A = make_wfragA(W, B, 8, 8);
  int lane = threadIdx.x & 31;
  int wave = (blockIdx.x * blockDim.x + threadIdx.x) >> 5;
  int nwaves = (gridDim.x * blockDim.x) >> 5;
  int ntiles = (N + 15) >> 4;
  float s[8], q[8];
#pragma unroll
  for (int r = 0; r < 8; ++r) { s[r] = 0.0f; q[r] = 0.0f; }
  for (int t = wave; t < ntiles; t += nwaves) {
    int n = t * 16 + (lane & 15);
    bool valid = (lane < 16) && (n < N);
    float f[8];
#pragma unroll
    for (int r = 0; r < 8; ++r) f[r] = 0.0f;
    if (valid) {
      float4 hn = ((const float4*)h)[n];
      float cc = fmaxf(cnt[n], 1.0f);
      f[0] = hn.x; f[1] = hn.y; f[2] = hn.z; f[3] = hn.w;
      f[4] = aggA[2 * n + 0];
      f[5] = aggA[2 * n + 1];
      f[6] = aggM[2 * n + 0] / cc;
      f[7] = aggM[2 * n + 1] / cc;
    }
    v16h Bf = make_bfrag(f, 8, valid);
    v8f y = wmma16(A, Bf);
    if (valid) {
#pragma unroll
      for (int r = 0; r < 8; ++r) {
        y1[n * 8 + r] = y[r];
        s[r] += y[r];
        q[r] += y[r] * y[r];
      }
    }
  }
  stats_reduce_store<8>(s, q, part);
}

// Node pass 2: y2 = tanh(bn3(y1)) @ upd_w2 + b2, store + stats (4 ch over N)
__global__ void __launch_bounds__(256)
node_pass2(const float* __restrict__ y1, int N, const float* __restrict__ W,
           const float* __restrict__ B, const float* __restrict__ bn3,
           float* __restrict__ y2, float* __restrict__ part) {
  v16h A = make_wfragA(W, B, 8, 4);
  float a3[8], c3[8];
#pragma unroll
  for (int r = 0; r < 8; ++r) { a3[r] = bn3[r]; c3[r] = bn3[8 + r]; }
  int lane = threadIdx.x & 31;
  int wave = (blockIdx.x * blockDim.x + threadIdx.x) >> 5;
  int nwaves = (gridDim.x * blockDim.x) >> 5;
  int ntiles = (N + 15) >> 4;
  float s[4], q[4];
#pragma unroll
  for (int r = 0; r < 4; ++r) { s[r] = 0.0f; q[r] = 0.0f; }
  for (int t = wave; t < ntiles; t += nwaves) {
    int n = t * 16 + (lane & 15);
    bool valid = (lane < 16) && (n < N);
    float f[8];
#pragma unroll
    for (int r = 0; r < 8; ++r) f[r] = 0.0f;
    if (valid) {
#pragma unroll
      for (int r = 0; r < 8; ++r) f[r] = y1[n * 8 + r];
    }
    float tv[8];
#pragma unroll
    for (int r = 0; r < 8; ++r) tv[r] = tanhf(a3[r] * f[r] + c3[r]);
    v16h Bf = make_bfrag(tv, 8, valid);
    v8f y = wmma16(A, Bf);
    if (valid) {
#pragma unroll
      for (int r = 0; r < 4; ++r) {
        y2[n * 4 + r] = y[r];
        s[r] += y[r];
        q[r] += y[r] * y[r];
      }
    }
  }
  stats_reduce_store<4>(s, q, part);
}

// Reduce block partials (f64), emit per-channel affine (a = g*rsqrt(var+eps),
// c = beta - mu*a) so BN becomes a*x + c in the consumer passes.
__global__ void finalize_bn(const float* __restrict__ part, int nblocks,
                            int nch, double count,
                            const float* __restrict__ gamma,
                            const float* __restrict__ beta,
                            float* __restrict__ ac) {
  int t = threadIdx.x;
  if (t < nch) {
    double s = 0.0, q = 0.0;
    for (int b = 0; b < nblocks; ++b) {
      s += (double)part[b * 2 * nch + t];
      q += (double)part[b * 2 * nch + nch + t];
    }
    double mu = s / count;
    double var = q / count - mu * mu;
    float a = gamma[t] * rsqrtf((float)var + 1e-5f);
    ac[t] = a;
    ac[nch + t] = beta[t] - (float)mu * a;
  }
}

__global__ void final_out(const float* __restrict__ y2, int N,
                          const float* __restrict__ bn4,
                          const float* __restrict__ pw,
                          const float* __restrict__ pb,
                          float* __restrict__ out) {
  int n = blockIdx.x * blockDim.x + threadIdx.x;
  if (n >= N) return;
  float o0 = pb[0], o1 = pb[1];
#pragma unroll
  for (int ch = 0; ch < 4; ++ch) {
    float u = tanhf(bn4[ch] * y2[n * 4 + ch] + bn4[4 + ch]);
    o0 += u * pw[ch * 2 + 0];
    o1 += u * pw[ch * 2 + 1];
  }
  out[n * 2 + 0] = o0;
  out[n * 2 + 1] = o1;
}

// ---------------------------------------------------------------------------
extern "C" void kernel_launch(void* const* d_in, const int* in_sizes, int n_in,
                              void* d_out, int out_size, void* d_ws,
                              size_t ws_size, hipStream_t stream) {
  const float* pos  = (const float*)d_in[0];
  const float* vel  = (const float*)d_in[1];
  const int*   ei   = (const int*)d_in[2];
  const float* lw   = (const float*)d_in[3];
  const float* lb   = (const float*)d_in[4];
  const float* mw1  = (const float*)d_in[5];
  const float* mb1  = (const float*)d_in[6];
  const float* mg1  = (const float*)d_in[7];
  const float* mbe1 = (const float*)d_in[8];
  const float* mw2  = (const float*)d_in[9];
  const float* mb2  = (const float*)d_in[10];
  const float* mg2  = (const float*)d_in[11];
  const float* mbe2 = (const float*)d_in[12];
  const float* uw1  = (const float*)d_in[13];
  const float* ub1  = (const float*)d_in[14];
  const float* ug1  = (const float*)d_in[15];
  const float* ube1 = (const float*)d_in[16];
  const float* uw2  = (const float*)d_in[17];
  const float* ub2  = (const float*)d_in[18];
  const float* ug2  = (const float*)d_in[19];
  const float* ube2 = (const float*)d_in[20];
  const float* pw   = (const float*)d_in[21];
  const float* pb   = (const float*)d_in[22];

  const int N = in_sizes[0] / 2;
  const int E = in_sizes[2] / 2;
  const int TB = 256;
  const int NBE = 1024;  // edge-pass blocks (8 waves each)
  const int NBN = 256;   // node-pass blocks

  float* ws   = (float*)d_ws;
  float* h    = ws;                       // 4N
  float* aggA = h + (size_t)4 * N;        // 2N
  float* aggM = aggA + (size_t)2 * N;     // 2N
  float* cnt  = aggM + (size_t)2 * N;     // N
  float* y1   = cnt + (size_t)N;          // 8N
  float* y2   = y1 + (size_t)8 * N;       // 4N
  float* bn   = y2 + (size_t)4 * N;       // 64: a1[8]c1[8] a2[4]c2[4] a3[8]c3[8] a4[4]c4[4]
  float* part1 = bn + 64;                 // NBE*16
  float* part2 = part1 + (size_t)NBE * 16;  // NBE*8
  float* part3 = part2 + (size_t)NBE * 8;   // NBN*16
  float* part4 = part3 + (size_t)NBN * 16;  // NBN*8

  zero_kernel<<<256, TB, 0, stream>>>(aggA, 5 * N);  // aggA|aggM|cnt contiguous
  compute_h<<<(N + TB - 1) / TB, TB, 0, stream>>>(pos, vel, N, lw, lb, h);

  edge_pass1<<<NBE, TB, 0, stream>>>(ei, h, E, mw1, mb1, part1);
  finalize_bn<<<1, 32, 0, stream>>>(part1, NBE, 8, (double)E, mg1, mbe1, bn + 0);
  edge_pass2<<<NBE, TB, 0, stream>>>(ei, h, E, mw1, mb1, mw2, mb2, bn + 0, part2);
  finalize_bn<<<1, 32, 0, stream>>>(part2, NBE, 4, (double)E, mg2, mbe2, bn + 16);
  edge_pass3<<<NBE, TB, 0, stream>>>(ei, h, E, mw1, mb1, mw2, mb2, bn + 0,
                                     bn + 16, aggA, aggM, cnt);

  node_pass1<<<NBN, TB, 0, stream>>>(h, aggA, aggM, cnt, N, uw1, ub1, y1, part3);
  finalize_bn<<<1, 32, 0, stream>>>(part3, NBN, 8, (double)N, ug1, ube1, bn + 24);
  node_pass2<<<NBN, TB, 0, stream>>>(y1, N, uw2, ub2, bn + 24, y2, part4);
  finalize_bn<<<1, 32, 0, stream>>>(part4, NBN, 4, (double)N, ug2, ube2, bn + 40);

  final_out<<<(N + TB - 1) / TB, TB, 0, stream>>>(y2, N, bn + 40, pw, pb,
                                                  (float*)d_out);
}
